// DigitCapsules_32959579030393
// MI455X (gfx1250) — compile-verified, matrix operand order checked by reading, not run on hardware
//
#include <hip/hip_runtime.h>
#include <hip/hip_bf16.h>

#define N_IN   1152
#define D_IN   8
#define N_OUT  10
#define D_OUT  16
#define BATCH  256
#define OJ     160   // N_OUT * D_OUT
#define RT_THREADS 320
#define NSTRIP 8
#define NPART  144   // N_IN / NSTRIP

typedef __attribute__((ext_vector_type(16))) __bf16 bf16x16;
typedef __attribute__((ext_vector_type(8)))  float  f32x8;

__device__ __forceinline__ float bf_lo(unsigned u) { return __uint_as_float(u << 16); }
__device__ __forceinline__ float bf_hi(unsigned u) { return __uint_as_float(u & 0xffff0000u); }

// ---------------------------------------------------------------------------
// Kernel 1: u_hat[b,n,o,j] = sum_i W[n,o,j,i] * x[b,n,i], stored bf16.
// One wave per (n, batch-tile-of-16). M=batch(16), N=j(16, one o per tile),
// K=i(8, zero-padded to 32) -> v_wmma_f32_16x16x32_bf16, 10 WMMAs per wave.
// All W loads and u_hat stores use one base address + immediate offsets
// (row stride 368640 B and o stride 32 B both fit the signed-24-bit IOFFSET).
// ---------------------------------------------------------------------------
__global__ __launch_bounds__(128) void uhat_wmma_kernel(
    const float* __restrict__ x, const float* __restrict__ W,
    __bf16* __restrict__ uhat)
{
    int wave = (blockIdx.x * 128 + threadIdx.x) >> 5;   // 0 .. 18431
    int lane = threadIdx.x & 31;
    int n  = wave >> 4;          // 0 .. 1151
    int bt = wave & 15;          // 0 .. 15
    int b0 = bt * 16;

    // A fragment: 16-bit A 16x32 layout. Lanes 0-15 hold K=0..7 (elements 0..7),
    // everything else (elements 8..15 on low lanes, all of high lanes) is zero padding.
    bf16x16 a = {};
    const float* wrbase = nullptr;
    if (lane < 16) {
        const float4* xr = (const float4*)(x + ((size_t)(b0 + lane) * N_IN + n) * D_IN);
        float4 p = xr[0], q = xr[1];
        a[0] = (__bf16)p.x; a[1] = (__bf16)p.y; a[2] = (__bf16)p.z; a[3] = (__bf16)p.w;
        a[4] = (__bf16)q.x; a[5] = (__bf16)q.y; a[6] = (__bf16)q.z; a[7] = (__bf16)q.w;
        wrbase = W + ((size_t)n * N_OUT * D_OUT + lane) * D_IN;  // + o*128 floats
    }

    int col   = lane & 15;
    int mbase = (lane >> 4) * 8;
    // One store base per wave; per-(r,o) deltas are compile-time immediates.
    __bf16* dst = uhat + ((size_t)(b0 + mbase) * N_IN + n) * OJ + col;

#pragma unroll
    for (int o = 0; o < N_OUT; ++o) {
        // B fragment: 32x16 bf16; lane c<16 holds B[K=0..7, c] in elements 0..7.
        // B[k, j] = W[n, o, j, k]; K=0..7 valid, rest zero.
        bf16x16 bm = {};
        if (lane < 16) {
            const float4* wr = (const float4*)(wrbase + (size_t)o * (D_OUT * D_IN));
            float4 p = wr[0], q = wr[1];
            bm[0] = (__bf16)p.x; bm[1] = (__bf16)p.y; bm[2] = (__bf16)p.z; bm[3] = (__bf16)p.w;
            bm[4] = (__bf16)q.x; bm[5] = (__bf16)q.y; bm[6] = (__bf16)q.z; bm[7] = (__bf16)q.w;
        }
        f32x8 c = {};
        c = __builtin_amdgcn_wmma_f32_16x16x32_bf16(
                /*neg_a=*/false, a, /*neg_b=*/false, bm,
                /*c_mod=*/(short)0, c, /*reuse_a=*/false, /*reuse_b=*/false);
        // C layout: VGPR r -> M = mbase + r, N = col.
#pragma unroll
        for (int r = 0; r < 8; ++r)
            dst[(size_t)r * ((size_t)N_IN * OJ) + o * D_OUT] = (__bf16)c[r];
    }
}

// ---------------------------------------------------------------------------
// Kernel 2: 3 routing iterations, one block per batch element.
// Pass k: [b += u.v_prev; c = softmax_o(b)]  -> [s = sum_n c*u]  -> [v = squash(s)]
// Pass 0 short-circuits to c = 0.1 (softmax of zeros), saving one u_hat sweep.
// ---------------------------------------------------------------------------
__global__ __launch_bounds__(RT_THREADS) void routing_kernel(
    const __bf16* __restrict__ uhat, float* __restrict__ bij_g,
    float* __restrict__ out)
{
    __shared__ float cij[N_IN * N_OUT];   // 46080 B
    __shared__ float ssh[NSTRIP * OJ];    //  5120 B
    __shared__ float ssum[OJ];
    __shared__ float vsh[OJ];

    int b   = blockIdx.x;
    int tid = threadIdx.x;
    const __bf16* U = uhat + (size_t)b * N_IN * OJ;
    float* bb = bij_g + (size_t)b * N_IN * N_OUT;

    int q     = tid % 40;        // (o, j-quad) slot for phase 2
    int strip = tid / 40;        // 0..7
    int oq    = q >> 2;
    int j4    = (q & 3) << 2;

    for (int pass = 0; pass < 3; ++pass) {
        // -------- phase 1: logits update + softmax -> cij --------
        if (pass == 0) {
            for (int i = tid; i < N_IN * N_OUT; i += RT_THREADS) cij[i] = 0.1f;
        } else {
            for (int n = tid; n < N_IN; n += RT_THREADS) {
                const __bf16* ur = U + (size_t)n * OJ;
                float bo[N_OUT];
                float eo[N_OUT];
#pragma unroll
                for (int o = 0; o < N_OUT; ++o) {
                    const uint4* up = (const uint4*)(ur + o * D_OUT);
                    uint4 u0 = up[0], u1 = up[1];
                    const float4* vp = (const float4*)(vsh + o * D_OUT);
                    float4 v0 = vp[0], v1 = vp[1], v2 = vp[2], v3 = vp[3];
                    float acc;
                    acc  = bf_lo(u0.x) * v0.x + bf_hi(u0.x) * v0.y;
                    acc += bf_lo(u0.y) * v0.z + bf_hi(u0.y) * v0.w;
                    acc += bf_lo(u0.z) * v1.x + bf_hi(u0.z) * v1.y;
                    acc += bf_lo(u0.w) * v1.z + bf_hi(u0.w) * v1.w;
                    acc += bf_lo(u1.x) * v2.x + bf_hi(u1.x) * v2.y;
                    acc += bf_lo(u1.y) * v2.z + bf_hi(u1.y) * v2.w;
                    acc += bf_lo(u1.z) * v3.x + bf_hi(u1.z) * v3.y;
                    acc += bf_lo(u1.w) * v3.z + bf_hi(u1.w) * v3.w;
                    float bprev = (pass == 1) ? 0.0f : bb[n * N_OUT + o];
                    float bn = bprev + acc;
                    if (pass == 1) bb[n * N_OUT + o] = bn;  // only pass 2 re-reads
                    bo[o] = bn;
                }
                float m = bo[0];
#pragma unroll
                for (int o = 1; o < N_OUT; ++o) m = fmaxf(m, bo[o]);
                float s = 0.0f;
#pragma unroll
                for (int o = 0; o < N_OUT; ++o) { eo[o] = __expf(bo[o] - m); s += eo[o]; }
                float inv = 1.0f / s;
#pragma unroll
                for (int o = 0; o < N_OUT; ++o) cij[n * N_OUT + o] = eo[o] * inv;
            }
        }
        __syncthreads();

        // -------- phase 2: s[o][j] = sum_n c[n][o] * u[n][o][j] (coalesced) ----
        {
            float a0 = 0.f, a1 = 0.f, a2 = 0.f, a3 = 0.f;
            const __bf16* base = U + (size_t)(strip * NPART) * OJ + oq * D_OUT + j4;
            const float* crow = cij + strip * NPART * N_OUT + oq;
            for (int i = 0; i < NPART; ++i) {
                uint2 w = *(const uint2*)(base + (size_t)i * OJ);
                float c = crow[i * N_OUT];
                a0 += c * bf_lo(w.x); a1 += c * bf_hi(w.x);
                a2 += c * bf_lo(w.y); a3 += c * bf_hi(w.y);
            }
            *(float4*)(ssh + strip * OJ + oq * D_OUT + j4) = make_float4(a0, a1, a2, a3);
        }
        __syncthreads();

        // -------- phase 3: strip reduce + squash -> vsh --------
        if (tid < OJ) {
            float s = 0.f;
#pragma unroll
            for (int st = 0; st < NSTRIP; ++st) s += ssh[st * OJ + tid];
            ssum[tid] = s;
        }
        __syncthreads();
        if (tid < OJ) {
            int o = tid >> 4;
            float sq = 0.f;
#pragma unroll
            for (int j = 0; j < D_OUT; ++j) { float t = ssum[o * D_OUT + j]; sq += t * t; }
            float s = ssum[tid];
            float v = (sq / (1.0f + sq)) * __frsqrt_rn(sq + 1e-8f) * s;
            vsh[tid] = v;
            if (pass == 2) out[(size_t)b * OJ + tid] = v;
        }
        __syncthreads();
    }
}

extern "C" void kernel_launch(void* const* d_in, const int* in_sizes, int n_in,
                              void* d_out, int out_size, void* d_ws, size_t ws_size,
                              hipStream_t stream) {
    (void)in_sizes; (void)n_in; (void)out_size; (void)ws_size;
    const float* x = (const float*)d_in[0];
    const float* W = (const float*)d_in[1];   // (1,1152,10,16,8) -> flat (1152,10,16,8)

    __bf16* uhat = (__bf16*)d_ws;                               // 94.4 MB
    size_t uhat_bytes = (size_t)BATCH * N_IN * OJ * sizeof(__bf16);
    float* bij = (float*)((char*)d_ws + uhat_bytes);            // +11.8 MB

    // 1152 n * 16 batch-tiles = 18432 waves, 4 waves (128 thr) per block
    uhat_wmma_kernel<<<dim3((N_IN * 16) / 4), dim3(128), 0, stream>>>(x, W, uhat);
    routing_kernel<<<dim3(BATCH), dim3(RT_THREADS), 0, stream>>>(uhat, bij, (float*)d_out);
}
